// ODE_RNN_24696061952557
// MI455X (gfx1250) — matrix operand order repaired and checked by reading, not compiled, not run on hardware
//
#include <hip/hip_runtime.h>
#include <hip/hip_bf16.h>

// ---------------------------------------------------------------------------
// ODE-RNN (GRU + RK4 neural-ODE between observations) for MI455X / gfx1250.
// B=256, T=64, H=256, D=512.  Persistent-state design: each workgroup owns a
// 16-row batch tile; h lives in LDS across all 64 time steps.  All GEMMs run
// on v_wmma_f32_16x16x32_f16 (f16 A/B, f32 accumulate).  GEMM inputs are kept
// as f16 LDS mirrors so the inner loops are pure ds_load_b32 + v_wmma, with
// k-outer / n-inner multi-accumulator structure (rolled k-loops: no spills).
// ---------------------------------------------------------------------------

typedef __attribute__((ext_vector_type(16))) _Float16 v16h;
typedef __attribute__((ext_vector_type(8)))  float    v8f;

#define B_DIM   256
#define T_STEPS 64
#define H_DIM   256
#define D_DIM   512
#define G_DIM   768   // 3*H
#define MT      16    // batch rows per workgroup
#define NSUB    2     // RK4 substeps per observation interval

#define V8F_ZERO {0.f, 0.f, 0.f, 0.f, 0.f, 0.f, 0.f, 0.f}

// ---- WMMA fragment loaders (CDNA5 wave32 layouts, CDNA5 ISA) ----

// A fragment (16x32) from an f16 row-major LDS tile (stride in halfs).
// lane L holds M=L%16; half slot s: K = ((L>=16)?8:0) + ((s>=8)?16:0) + (s&7).
__device__ __forceinline__ v16h load_a_f16(const _Float16* buf, int stride,
                                           int k0, int lane) {
    union { v16h v; unsigned u32[8]; } u;
    const _Float16* row = buf + (lane & 15) * stride + k0 + ((lane >> 4) << 3);
#pragma unroll
    for (int i = 0; i < 8; ++i) {
        int s = 2 * i;
        int k = (s & 7) + ((s & 8) << 1);   // consecutive K pairs
        u.u32[i] = *(const unsigned*)(row + k);
    }
    return u.v;
}

// B fragment (32x16) = 16 rows of a row-major weight matrix W[n, kdim].
// lane L holds N=L%16; half slot s: K = k0 + ((L>=16)?8:0) + ((s>=8)?16:0) + (s&7).
__device__ __forceinline__ v16h load_b(const _Float16* __restrict__ W, int kdim,
                                       int n0, int k0, int lane) {
    const _Float16* p =
        W + (size_t)(n0 + (lane & 15)) * kdim + k0 + ((lane >> 4) << 3);
    union { v16h v; uint4 q[2]; } u;
    u.q[0] = *(const uint4*)p;
    u.q[1] = *(const uint4*)(p + 16);
    return u.v;
}

__device__ __forceinline__ v8f wmma16(v16h a, v16h b, v8f c) {
    return __builtin_amdgcn_wmma_f32_16x16x32_f16(
        /*neg_a=*/false, a, /*neg_b=*/false, b,
        /*c_mod=*/(short)0, c, /*reuse_a=*/false, /*reuse_b=*/false);
}

__device__ __forceinline__ float sigmoidf_dev(float x) {
    return 1.0f / (1.0f + __expf(-x));
}

// ---- C-tile store helpers (C/D layout: lane = N + 16*(M>=8), vgpr r = M%8) --
__device__ __forceinline__ void store_c_f16(_Float16* dst, int stride, int n,
                                            int mb, v8f acc, float bias) {
#pragma unroll
    for (int r = 0; r < 8; ++r)
        dst[(mb + r) * stride + n] = (_Float16)(acc[r] + bias);
}
__device__ __forceinline__ void store_c_tanh(float* dst, int stride, int n,
                                             int mb, v8f acc, float bias) {
#pragma unroll
    for (int r = 0; r < 8; ++r)
        dst[(mb + r) * stride + n] = tanhf(acc[r] + bias);
}
__device__ __forceinline__ void store_c_f32(float* dst, int stride, int n,
                                            int mb, v8f acc, float bias) {
#pragma unroll
    for (int r = 0; r < 8; ++r)
        dst[(mb + r) * stride + n] = acc[r] + bias;
}

// ---- one ODE RHS evaluation: out = tanh((in @ W1^T + b1) @ W2^T + b2) -----
// in_h: 16xH f16 LDS tile.  out: 16xH f32 LDS tile.  sh_z: 16xD f16 scratch.
__device__ __forceinline__ void mlp_eval(
    const _Float16* in_h, float* out, _Float16* sh_z,
    const _Float16* __restrict__ W1h, const float* __restrict__ b1,
    const _Float16* __restrict__ W2h, const float* __restrict__ b2,
    int wave, int lane) {
    const int nl = lane & 15;
    const int mb = (lane >> 4) << 3;

    // Stage 1: z = in @ W1^T + b1   (16 x 512): 4 N-tiles per wave, K=256.
    {
        v8f a0 = V8F_ZERO, a1 = V8F_ZERO, a2 = V8F_ZERO, a3 = V8F_ZERO;
#pragma clang loop unroll(disable)
        for (int kt = 0; kt < H_DIM / 32; ++kt) {
            v16h a = load_a_f16(in_h, H_DIM, kt * 32, lane);
            a0 = wmma16(a, load_b(W1h, H_DIM, (wave +  0) * 16, kt * 32, lane), a0);
            a1 = wmma16(a, load_b(W1h, H_DIM, (wave +  8) * 16, kt * 32, lane), a1);
            a2 = wmma16(a, load_b(W1h, H_DIM, (wave + 16) * 16, kt * 32, lane), a2);
            a3 = wmma16(a, load_b(W1h, H_DIM, (wave + 24) * 16, kt * 32, lane), a3);
        }
        store_c_f16(sh_z, D_DIM, (wave +  0) * 16 + nl, mb, a0, b1[(wave +  0) * 16 + nl]);
        store_c_f16(sh_z, D_DIM, (wave +  8) * 16 + nl, mb, a1, b1[(wave +  8) * 16 + nl]);
        store_c_f16(sh_z, D_DIM, (wave + 16) * 16 + nl, mb, a2, b1[(wave + 16) * 16 + nl]);
        store_c_f16(sh_z, D_DIM, (wave + 24) * 16 + nl, mb, a3, b1[(wave + 24) * 16 + nl]);
    }
    __syncthreads();

    // Stage 2: out = tanh(z @ W2^T + b2) (16 x 256): 2 N-tiles per wave, K=512.
    {
        v8f a0 = V8F_ZERO, a1 = V8F_ZERO;
#pragma clang loop unroll(disable)
        for (int kt = 0; kt < D_DIM / 32; ++kt) {
            v16h a = load_a_f16(sh_z, D_DIM, kt * 32, lane);
            a0 = wmma16(a, load_b(W2h, D_DIM, (wave + 0) * 16, kt * 32, lane), a0);
            a1 = wmma16(a, load_b(W2h, D_DIM, (wave + 8) * 16, kt * 32, lane), a1);
        }
        store_c_tanh(out, H_DIM, (wave + 0) * 16 + nl, mb, a0, b2[(wave + 0) * 16 + nl]);
        store_c_tanh(out, H_DIM, (wave + 8) * 16 + nl, mb, a1, b2[(wave + 8) * 16 + nl]);
    }
    __syncthreads();
}

// ---------------------------- weight conversion ----------------------------
__global__ void cvt_f32_to_f16_kernel(const float* __restrict__ src,
                                      _Float16* __restrict__ dst, int n) {
    int i = blockIdx.x * blockDim.x + threadIdx.x;
    if (i < n) dst[i] = (_Float16)src[i];
}

// ------------------------------- main kernel -------------------------------
__global__ void __launch_bounds__(256, 1) ode_rnn_persistent_kernel(
    const float* __restrict__ batch,  // (B, T, 2)
    const float* __restrict__ mask,   // (B, T)
    const float* __restrict__ b1, const float* __restrict__ b2,
    const float* __restrict__ W_ih,   // (3H, 1) flat
    const float* __restrict__ b_ih, const float* __restrict__ b_hh,
    const _Float16* __restrict__ W1h,   // (D, H) f16
    const _Float16* __restrict__ W2h,   // (H, D) f16
    const _Float16* __restrict__ Whh_h, // (3H, H) f16
    float* __restrict__ out)            // (B, H)
{
    extern __shared__ unsigned char smem[];
    // Layout (bytes):
    //   [0      ) h     f32 16KB   persistent state
    //   [16384  ) hf    f16  8KB   f16 mirror of h (GEMM A operand)
    //   [24576  ) tmpf  f16  8KB   RK stage input (f16 only)
    //   [32768  ) acc   f32 16KB \
    //   [49152  ) k     f32 16KB  } gh (16x768 f32, 48KB) aliases acc..z
    //   [65536  ) z     f16 16KB /
    float*    sh_h    = (float*)smem;
    _Float16* sh_hf   = (_Float16*)(smem + 16384);
    _Float16* sh_tmpf = (_Float16*)(smem + 24576);
    float*    sh_acc  = (float*)(smem + 32768);
    float*    sh_k    = (float*)(smem + 49152);
    _Float16* sh_z    = (_Float16*)(smem + 65536);
    float*    sh_gh   = sh_acc;   // GRU gates, used only when acc/k/z idle

    const int tid  = threadIdx.x;
    const int lane = tid & 31;
    const int wave = tid >> 5;
    const int rowbase = blockIdx.x * MT;

    for (int e = tid; e < MT * H_DIM; e += 256) {
        sh_h[e]  = 0.0f;
        sh_hf[e] = (_Float16)0.0f;
    }
    __syncthreads();

    for (int t = 0; t < T_STEPS; ++t) {
        const float t1v = batch[t * 2];
        const float t0v = (t == 0) ? 0.0f : batch[(t - 1) * 2];
        const float hs  = (t1v - t0v) * (1.0f / (float)NSUB);

        // ---- ODE integrate h over [t0, t1] with NSUB RK4 substeps ----
        for (int sub = 0; sub < NSUB; ++sub) {
            // k1 = f(h)
            mlp_eval(sh_hf, sh_k, sh_z, W1h, b1, W2h, b2, wave, lane);
            for (int e = tid; e < MT * H_DIM; e += 256) {
                float hv = sh_h[e], kv = sh_k[e];
                sh_acc[e]  = hv + (hs * (1.0f / 6.0f)) * kv;
                sh_tmpf[e] = (_Float16)(hv + 0.5f * hs * kv);
            }
            __syncthreads();
            // k2 = f(h + hs/2 k1)
            mlp_eval(sh_tmpf, sh_k, sh_z, W1h, b1, W2h, b2, wave, lane);
            for (int e = tid; e < MT * H_DIM; e += 256) {
                float kv = sh_k[e];
                sh_acc[e] += (hs * (1.0f / 3.0f)) * kv;
                sh_tmpf[e] = (_Float16)(sh_h[e] + 0.5f * hs * kv);
            }
            __syncthreads();
            // k3 = f(h + hs/2 k2)
            mlp_eval(sh_tmpf, sh_k, sh_z, W1h, b1, W2h, b2, wave, lane);
            for (int e = tid; e < MT * H_DIM; e += 256) {
                float kv = sh_k[e];
                sh_acc[e] += (hs * (1.0f / 3.0f)) * kv;
                sh_tmpf[e] = (_Float16)(sh_h[e] + hs * kv);
            }
            __syncthreads();
            // k4 = f(h + hs k3);  h += hs/6 (k1 + 2k2 + 2k3 + k4)
            mlp_eval(sh_tmpf, sh_k, sh_z, W1h, b1, W2h, b2, wave, lane);
            for (int e = tid; e < MT * H_DIM; e += 256) {
                float hv = sh_acc[e] + (hs * (1.0f / 6.0f)) * sh_k[e];
                sh_h[e]  = hv;
                sh_hf[e] = (_Float16)hv;
            }
            __syncthreads();
        }

        // ---- GRU gate GEMM: gh = hp @ W_hh^T + b_hh (16 x 768): 6 N-tiles ----
        {
            const int nl = lane & 15;
            const int mb = (lane >> 4) << 3;
            v8f a0 = V8F_ZERO, a1 = V8F_ZERO, a2 = V8F_ZERO;
            v8f a3 = V8F_ZERO, a4 = V8F_ZERO, a5 = V8F_ZERO;
#pragma clang loop unroll(disable)
            for (int kt = 0; kt < H_DIM / 32; ++kt) {
                v16h a = load_a_f16(sh_hf, H_DIM, kt * 32, lane);
                a0 = wmma16(a, load_b(Whh_h, H_DIM, (wave +  0) * 16, kt * 32, lane), a0);
                a1 = wmma16(a, load_b(Whh_h, H_DIM, (wave +  8) * 16, kt * 32, lane), a1);
                a2 = wmma16(a, load_b(Whh_h, H_DIM, (wave + 16) * 16, kt * 32, lane), a2);
                a3 = wmma16(a, load_b(Whh_h, H_DIM, (wave + 24) * 16, kt * 32, lane), a3);
                a4 = wmma16(a, load_b(Whh_h, H_DIM, (wave + 32) * 16, kt * 32, lane), a4);
                a5 = wmma16(a, load_b(Whh_h, H_DIM, (wave + 40) * 16, kt * 32, lane), a5);
            }
            store_c_f32(sh_gh, G_DIM, (wave +  0) * 16 + nl, mb, a0, b_hh[(wave +  0) * 16 + nl]);
            store_c_f32(sh_gh, G_DIM, (wave +  8) * 16 + nl, mb, a1, b_hh[(wave +  8) * 16 + nl]);
            store_c_f32(sh_gh, G_DIM, (wave + 16) * 16 + nl, mb, a2, b_hh[(wave + 16) * 16 + nl]);
            store_c_f32(sh_gh, G_DIM, (wave + 24) * 16 + nl, mb, a3, b_hh[(wave + 24) * 16 + nl]);
            store_c_f32(sh_gh, G_DIM, (wave + 32) * 16 + nl, mb, a4, b_hh[(wave + 32) * 16 + nl]);
            store_c_f32(sh_gh, G_DIM, (wave + 40) * 16 + nl, mb, a5, b_hh[(wave + 40) * 16 + nl]);
        }
        __syncthreads();

        // ---- GRU elementwise combine + mask blend (in-place on sh_h) ----
        for (int e = tid; e < MT * H_DIM; e += 256) {
            int m = e >> 8;       // H_DIM == 256
            int j = e & (H_DIM - 1);
            size_t bt = (size_t)(rowbase + m) * T_STEPS + t;
            float x  = batch[bt * 2 + 1];
            float mk = mask[bt];
            float hp = sh_h[e];
            float gir = x * W_ih[j]             + b_ih[j];
            float giz = x * W_ih[H_DIM + j]     + b_ih[H_DIM + j];
            float gin = x * W_ih[2 * H_DIM + j] + b_ih[2 * H_DIM + j];
            float ghr = sh_gh[m * G_DIM + j];
            float ghz = sh_gh[m * G_DIM + H_DIM + j];
            float ghn = sh_gh[m * G_DIM + 2 * H_DIM + j];
            float r  = sigmoidf_dev(gir + ghr);
            float zz = sigmoidf_dev(giz + ghz);
            float nn = tanhf(gin + r * ghn);
            float ht = (1.0f - zz) * nn + zz * hp;
            float hnew = mk * ht + (1.0f - mk) * hp;
            sh_h[e]  = hnew;
            sh_hf[e] = (_Float16)hnew;
        }
        __syncthreads();
    }

    // ---- write final state ----
    for (int e = tid; e < MT * H_DIM; e += 256)
        out[(size_t)(rowbase + (e >> 8)) * H_DIM + (e & (H_DIM - 1))] = sh_h[e];
}

// ------------------------------- launcher ----------------------------------
extern "C" void kernel_launch(void* const* d_in, const int* in_sizes, int n_in,
                              void* d_out, int out_size, void* d_ws, size_t ws_size,
                              hipStream_t stream) {
    const float* batch = (const float*)d_in[0];
    const float* mask  = (const float*)d_in[1];
    const float* W1    = (const float*)d_in[2];
    const float* b1    = (const float*)d_in[3];
    const float* W2    = (const float*)d_in[4];
    const float* b2    = (const float*)d_in[5];
    const float* W_ih  = (const float*)d_in[6];
    const float* b_ih  = (const float*)d_in[7];
    const float* W_hh  = (const float*)d_in[8];
    const float* b_hh  = (const float*)d_in[9];
    float* out = (float*)d_out;
    (void)in_sizes; (void)n_in; (void)out_size; (void)ws_size;

    // f16 weight copies in workspace (L2-resident thereafter).
    _Float16* W1h = (_Float16*)d_ws;                 // D*H   = 131072 halfs
    _Float16* W2h = W1h + (size_t)D_DIM * H_DIM;     // H*D   = 131072 halfs
    _Float16* Whh = W2h + (size_t)H_DIM * D_DIM;     // 3H*H  = 196608 halfs

    const int n1 = D_DIM * H_DIM, n2 = H_DIM * D_DIM, n3 = G_DIM * H_DIM;
    cvt_f32_to_f16_kernel<<<(n1 + 255) / 256, 256, 0, stream>>>(W1, W1h, n1);
    cvt_f32_to_f16_kernel<<<(n2 + 255) / 256, 256, 0, stream>>>(W2, W2h, n2);
    cvt_f32_to_f16_kernel<<<(n3 + 255) / 256, 256, 0, stream>>>(W_hh, Whh, n3);

    const size_t smem_bytes = 81920;  // see layout comment in kernel

    ode_rnn_persistent_kernel<<<B_DIM / MT, 256, smem_bytes, stream>>>(
        batch, mask, b1, b2, W_ih, b_ih, b_hh, W1h, W2h, Whh, out);
}
